// Vanilla_VectorQuantizer_87952340287716
// MI455X (gfx1250) — matrix operand order repaired and checked by reading, not compiled, not run on hardware
//
#include <hip/hip_runtime.h>

#define N_TOKENS   131072
#define EMBED_DIM  128
#define NUM_EMBED  2048
#define KP         64                  // packed bf16-pairs per row (128/2)
#define CHUNK_NT   4                   // n-tiles per LDS chunk (64 codes)
#define CHUNK_CODES (CHUNK_NT * 16)    // 64
#define CHUNK_WORDS (CHUNK_CODES * KP) // 4096 u32 = 16 KB
#define NCHUNKS    (NUM_EMBED / CHUNK_CODES) // 32
#define WG_TOKENS  128
#define NBLOCKS    (N_TOKENS / WG_TOKENS)    // 1024
#define SLICE_WORDS (CHUNK_WORDS / 8)  // 512 u32 per wave per buffer

typedef __attribute__((ext_vector_type(16))) __bf16        v16bf;
typedef __attribute__((ext_vector_type(8)))  float         v8f;
typedef __attribute__((ext_vector_type(4)))  unsigned int  u32x4;
typedef __attribute__((ext_vector_type(4)))  int           i32x4;
typedef __attribute__((ext_vector_type(8)))  int           i32x8;
typedef __attribute__((ext_vector_type(4)))  float         f32x4;

union BF16Frag {
  unsigned short u[16];
  unsigned int   w[8];
  u32x4          q[2];
  v16bf          v;
};

__device__ __forceinline__ unsigned short f2bf_rne(float f) {
  unsigned int u = __float_as_uint(f);
  unsigned int r = u + 0x7FFFu + ((u >> 16) & 1u);
  return (unsigned short)(r >> 16);
}
__device__ __forceinline__ float bf2f(unsigned short h) {
  return __uint_as_float(((unsigned int)h) << 16);
}

// ---------------------------------------------------------------------------
// TDM: 1-row tile of `nwords` dwords, global -> LDS, per ISA ch.8 D# layout.
// group0: [1:0]=count(1) [63:32]=lds_addr [120:64]=global_addr [127:126]=type(2)
// group1: [17:16]=data_size(2: 4B) [79:48]=tensor_dim0 [111:80]=tensor_dim1
//         [127:112]=tile_dim0 [143:128]=tile_dim1 [207:160]=dim0_stride
//         [255:208]=dim1_stride
// This toolchain exposes the 6-arg builtin: (g0, g1, g2, g3, g4, cpol).
// ---------------------------------------------------------------------------
__device__ __forceinline__ void tdm_load_1d(unsigned int lds_byte_addr,
                                            const unsigned int* __restrict__ gsrc,
                                            unsigned int nwords) {
  unsigned long long ga = (unsigned long long)gsrc;
  u32x4 g0;
  g0.x = 1u;                                            // count=1, user desc
  g0.y = lds_byte_addr;
  g0.z = (unsigned int)ga;
  g0.w = (unsigned int)((ga >> 32) & 0x1FFFFFFull) | (2u << 30);  // type=2
  i32x8 g1;
  g1[0] = (int)(2u << 16);                              // data_size = 4 bytes
  g1[1] = (int)((nwords & 0xFFFFu) << 16);              // tensor_dim0 lo16
  g1[2] = (int)((nwords >> 16) | (1u << 16));           // tdim0 hi16 | tensor_dim1=1 lo
  g1[3] = (int)((nwords & 0xFFFFu) << 16);              // tdim1 hi=0 | tile_dim0
  g1[4] = 1;                                            // tile_dim1=1, tile_dim2=0
  g1[5] = (int)nwords;                                  // dim0_stride lo32
  g1[6] = (int)((nwords & 0xFFFFu) << 16);              // dim0_stride hi | dim1_stride lo
  g1[7] = 0;                                            // dim1_stride hi
  i32x4 z4 = {0, 0, 0, 0};
  i32x8 z8 = {0, 0, 0, 0, 0, 0, 0, 0};
  __builtin_amdgcn_tensor_load_to_lds(g0, g1, z4, z4, z8, 0);
}

// ---------------------------------------------------------------------------
// Prep: pack codebook rows as bf16 hi/lo pairs ([code][kp] u32) + row norms.
// ---------------------------------------------------------------------------
__global__ void vq_prep(const float* __restrict__ cb,
                        unsigned int* __restrict__ cbp_hi,
                        unsigned int* __restrict__ cbp_lo,
                        float* __restrict__ cnorm) {
  int code = blockIdx.x * blockDim.x + threadIdx.x;
  if (code >= NUM_EMBED) return;
  const float* row = cb + code * EMBED_DIM;
  float s = 0.f;
  for (int kp = 0; kp < KP; ++kp) {
    float x0 = row[2 * kp + 0];
    float x1 = row[2 * kp + 1];
    s += x0 * x0 + x1 * x1;
    unsigned short h0 = f2bf_rne(x0), h1 = f2bf_rne(x1);
    unsigned short l0 = f2bf_rne(x0 - bf2f(h0));
    unsigned short l1 = f2bf_rne(x1 - bf2f(h1));
    cbp_hi[code * KP + kp] = (unsigned int)h0 | ((unsigned int)h1 << 16);
    cbp_lo[code * KP + kp] = (unsigned int)l0 | ((unsigned int)l1 << 16);
  }
  cnorm[code] = s;
}

// ---------------------------------------------------------------------------
// Main: fused distance GEMM (bf16x3 WMMA) + argmin + gather + loss partials.
// Block = 256 threads (8 waves); wave owns 16 tokens; codebook double-buffered
// through LDS by the Tensor Data Mover (TENSORcnt pipeline).
// ---------------------------------------------------------------------------
__launch_bounds__(256, 2)
__global__ void vq_main(const float* __restrict__ z,
                        const float* __restrict__ cb,
                        const unsigned int* __restrict__ cbp_hi,
                        const unsigned int* __restrict__ cbp_lo,
                        const float* __restrict__ cnorm,
                        float* __restrict__ out,        // [0]=loss slot, +1 quantized
                        float* __restrict__ partial) {
  __shared__ unsigned int sh_hi[2][CHUNK_WORDS];        // 2 x 16 KB
  __shared__ unsigned int sh_lo[2][CHUNK_WORDS];        // 2 x 16 KB

  const int tid   = threadIdx.x;
  const int lane  = tid & 31;
  const int wv    = tid >> 5;            // wave in block: 0..7
  const int half  = (lane >> 4) & 1;     // lane group
  const int lml   = lane & 15;
  const int mbase = blockIdx.x * WG_TOKENS + wv * 16;
  const int slice = wv * SLICE_WORDS;    // this wave's DMA slice within a chunk

  const unsigned int hi_base =
      (unsigned int)(unsigned long long)(const void*)&sh_hi[0][0];
  const unsigned int lo_base =
      (unsigned int)(unsigned long long)(const void*)&sh_lo[0][0];

  // ---- prologue: DMA chunk 0 into buffer 0 (each wave its own 2 KB slice)
  tdm_load_1d(hi_base + (unsigned)slice * 4u, cbp_hi + slice, SLICE_WORDS);
  tdm_load_1d(lo_base + (unsigned)slice * 4u, cbp_lo + slice, SLICE_WORDS);

  // ---- A fragments: 16-token tile of z, split to bf16 hi/lo, kept in regs.
  // A 16x32 bf16 layout: lane -> M=lane%16; VGPR v holds K = (v/4)*16 + half*8 + (v%4)*2 (+kk*32)
  BF16Frag a_hi[4], a_lo[4];
  {
    const float* zrow = z + (size_t)(mbase + lml) * EMBED_DIM;
#pragma unroll
    for (int kk = 0; kk < 4; ++kk) {
#pragma unroll
      for (int v = 0; v < 8; ++v) {
        int kp = kk * 16 + (v >> 2) * 8 + half * 4 + (v & 3);
        float x0 = zrow[2 * kp + 0];
        float x1 = zrow[2 * kp + 1];
        unsigned short h0 = f2bf_rne(x0), h1 = f2bf_rne(x1);
        a_hi[kk].u[2 * v + 0] = h0;
        a_hi[kk].u[2 * v + 1] = h1;
        a_lo[kk].u[2 * v + 0] = f2bf_rne(x0 - bf2f(h0));
        a_lo[kk].u[2 * v + 1] = f2bf_rne(x1 - bf2f(h1));
      }
    }
  }

  float bestd[8];
  int   bestn[8];
#pragma unroll
  for (int v = 0; v < 8; ++v) { bestd[v] = 3.4e38f; bestn[v] = 0; }

  for (int c = 0; c < NCHUNKS; ++c) {
    // my DMA ops for chunk c have landed; barrier makes all slices visible
    __builtin_amdgcn_s_wait_tensorcnt(0);
    __syncthreads();

    const int buf = c & 1;
    if (c + 1 < NCHUNKS) {  // overlap: DMA next chunk into the other buffer
      const int nb = (c + 1) & 1;
      tdm_load_1d(hi_base + (unsigned)(nb * CHUNK_WORDS + slice) * 4u,
                  cbp_hi + (c + 1) * CHUNK_WORDS + slice, SLICE_WORDS);
      tdm_load_1d(lo_base + (unsigned)(nb * CHUNK_WORDS + slice) * 4u,
                  cbp_lo + (c + 1) * CHUNK_WORDS + slice, SLICE_WORDS);
    }

#pragma unroll
    for (int t = 0; t < CHUNK_NT; ++t) {
      const int n_g = (c * CHUNK_NT + t) * 16 + lml;   // this lane's column
      const float cn = cnorm[n_g];
      v8f accA = {}, accB = {}, accC = {};             // 3 chains for XDL ILP
#pragma unroll
      for (int kk = 0; kk < 4; ++kk) {
        // B 32x16 bf16 layout: lane -> N=lane%16; VGPR v holds K = (v/4)*16 + half*8 + (v%4)*2 (+kk*32)
        BF16Frag b_hi, b_lo;
        const int idx = (t * 16 + lml) * KP + kk * 16 + half * 4;
        b_hi.q[0] = *(const u32x4*)&sh_hi[buf][idx + 0];
        b_hi.q[1] = *(const u32x4*)&sh_hi[buf][idx + 8];
        b_lo.q[0] = *(const u32x4*)&sh_lo[buf][idx + 0];
        b_lo.q[1] = *(const u32x4*)&sh_lo[buf][idx + 8];
        accA = __builtin_amdgcn_wmma_f32_16x16x32_bf16(false, a_hi[kk].v, false, b_hi.v,
                                                       (short)0, accA, false, false);
        accB = __builtin_amdgcn_wmma_f32_16x16x32_bf16(false, a_hi[kk].v, false, b_lo.v,
                                                       (short)0, accB, false, false);
        accC = __builtin_amdgcn_wmma_f32_16x16x32_bf16(false, a_lo[kk].v, false, b_hi.v,
                                                       (short)0, accC, false, false);
      }
      // dist = ||c||^2 - 2 z.c   (||z||^2 constant per row -> irrelevant to argmin)
#pragma unroll
      for (int v = 0; v < 8; ++v) {
        float d = cn - 2.0f * (accA[v] + accB[v] + accC[v]);
        if (d < bestd[v]) { bestd[v] = d; bestn[v] = n_g; }
      }
    }
  }

  // argmin across the 16 lanes of each half-group (ties -> lowest index)
#pragma unroll
  for (int v = 0; v < 8; ++v) {
    float d = bestd[v];
    int   n = bestn[v];
#pragma unroll
    for (int off = 1; off <= 8; off <<= 1) {
      float d2 = __shfl_xor(d, off, 32);
      int   n2 = __shfl_xor(n, off, 32);
      if (d2 < d || (d2 == d && n2 < n)) { d = d2; n = n2; }
    }
    bestn[v] = n;
  }

  // gather nearest codebook rows -> quantized output; accumulate (q-z)^2
  float lsum = 0.f;
  float* outq = out + 1;
#pragma unroll
  for (int v = 0; v < 8; ++v) {
    const int m = mbase + v + 8 * half;          // C/D layout row
    const f32x4* crow = (const f32x4*)(cb + (size_t)bestn[v] * EMBED_DIM);
    const f32x4* zr   = (const f32x4*)(z  + (size_t)m * EMBED_DIM);
    f32x4 q0 = crow[lml * 2 + 0], q1 = crow[lml * 2 + 1];
    f32x4 z0 = zr[lml * 2 + 0],   z1 = zr[lml * 2 + 1];
    float* orow = outq + (size_t)m * EMBED_DIM + lml * 8;
    orow[0] = q0.x; orow[1] = q0.y; orow[2] = q0.z; orow[3] = q0.w;
    orow[4] = q1.x; orow[5] = q1.y; orow[6] = q1.z; orow[7] = q1.w;
    float dx;
    dx = q0.x - z0.x; lsum += dx * dx;
    dx = q0.y - z0.y; lsum += dx * dx;
    dx = q0.z - z0.z; lsum += dx * dx;
    dx = q0.w - z0.w; lsum += dx * dx;
    dx = q1.x - z1.x; lsum += dx * dx;
    dx = q1.y - z1.y; lsum += dx * dx;
    dx = q1.z - z1.z; lsum += dx * dx;
    dx = q1.w - z1.w; lsum += dx * dx;
  }

  // deterministic block-level loss partial (reuse LDS)
  __syncthreads();
  float* red = (float*)&sh_hi[0][0];
  red[tid] = lsum;
  __syncthreads();
  for (int s = 128; s > 0; s >>= 1) {
    if (tid < s) red[tid] += red[tid + s];
    __syncthreads();
  }
  if (tid == 0) partial[blockIdx.x] = red[0];
}

// ---------------------------------------------------------------------------
// Deterministic final loss reduction (1024 partials -> out[0])
// ---------------------------------------------------------------------------
__global__ void vq_reduce(const float* __restrict__ partial, float* __restrict__ out) {
  __shared__ float red[256];
  int tid = threadIdx.x;
  float s = partial[tid] + partial[tid + 256] + partial[tid + 512] + partial[tid + 768];
  red[tid] = s;
  __syncthreads();
  for (int k = 128; k > 0; k >>= 1) {
    if (tid < k) red[tid] += red[tid + k];
    __syncthreads();
  }
  if (tid == 0) out[0] = red[0] * (1.0f / (float)(N_TOKENS * EMBED_DIM));
}

extern "C" void kernel_launch(void* const* d_in, const int* in_sizes, int n_in,
                              void* d_out, int out_size, void* d_ws, size_t ws_size,
                              hipStream_t stream) {
  (void)in_sizes; (void)n_in; (void)out_size; (void)ws_size;
  const float* z  = (const float*)d_in[0];
  const float* cb = (const float*)d_in[1];

  unsigned int* cbp_hi = (unsigned int*)d_ws;                 // 512 KB
  unsigned int* cbp_lo = cbp_hi + (size_t)NUM_EMBED * KP;     // 512 KB
  float* cnorm   = (float*)(cbp_lo + (size_t)NUM_EMBED * KP); // 8 KB
  float* partial = cnorm + NUM_EMBED;                         // 4 KB
  float* out = (float*)d_out;

  vq_prep  <<<NUM_EMBED / 256, 256, 0, stream>>>(cb, cbp_hi, cbp_lo, cnorm);
  vq_main  <<<NBLOCKS,        256, 0, stream>>>(z, cb, cbp_hi, cbp_lo, cnorm, out, partial);
  vq_reduce<<<1,              256, 0, stream>>>(partial, out);
}
